// GatedAttentionBasedRNN_1297080124080
// MI455X (gfx1250) — matrix operand order repaired and checked by reading, not compiled, not run on hardware
//
#include <hip/hip_runtime.h>
#include <math.h>

// ---------------- types ----------------
typedef __attribute__((ext_vector_type(16))) __bf16 v16bf;
typedef __attribute__((ext_vector_type(8)))  float  v8f;

// Problem constants
#define PP 512
#define QQ 64
#define BB 16
#define EE 256
#define HH 256
#define OO 256

// ---------------- WMMA helpers (CDNA5 16x16x32 bf16) ----------------
__device__ __forceinline__ v8f wmma_bf16(v16bf a, v16bf b, v8f c) {
    return __builtin_amdgcn_wmma_f32_16x16x32_bf16(
        /*neg_a=*/false, a, /*neg_b=*/false, b,
        /*c_mod=*/(short)0, c, /*reuse_a=*/false, /*reuse_b=*/false);
}

// A fragment: 16x32 (MxK) bf16, A row-major with leading dim lda.
// Lane L: m = L%16, half = L/16.
// element e holds K = half*16 + e
__device__ __forceinline__ v16bf load_frag_A(const __bf16* __restrict__ A,
                                             int lda, int m0, int k0) {
    int lane = threadIdx.x & 31;
    int m = lane & 15, half = (lane >> 4) & 1;
    const __bf16* base = A + (size_t)(m0 + m) * lda + k0;
    v16bf f;
#pragma unroll
    for (int i = 0; i < 16; ++i) f[i] = base[half * 16 + i];
    return f;
}

// B fragment: 32x16 (KxN), with B[k][n] = W[n0+n][k0+k], W row-major (N x K), ld ldw.
// Lane L: n = L%16, half = L/16; element e holds K = half*16 + e
// -> 16 contiguous bf16 from a W row (32-byte load).
__device__ __forceinline__ v16bf load_frag_B(const __bf16* __restrict__ W,
                                             int ldw, int n0, int k0) {
    int lane = threadIdx.x & 31;
    int n = lane & 15, half = (lane >> 4) & 1;
    const __bf16* base = W + (size_t)(n0 + n) * ldw + k0 + half * 16;
    v16bf f;
#pragma unroll
    for (int i = 0; i < 16; ++i) f[i] = base[i];
    return f;
}

// ---------------- f32 -> bf16 conversion ----------------
__global__ void f32_to_bf16_kernel(const float* __restrict__ in,
                                   __bf16* __restrict__ out, int n) {
    int i = blockIdx.x * blockDim.x + threadIdx.x;
    if (i < n) out[i] = (__bf16)in[i];
}

// ---------------- generic WMMA GEMM: D = A(MxK) * W(NxK)^T ----------------
// Each wave computes one 16x64 strip: 1 M-tile x 4 N-tiles.
// A fragment reused across the 4 N-tiles; 4 independent accumulator chains.
// mode 0: outf[m*N+n] = d
// mode 1: outf[m*N+n] = d + bias[n]
// mode 2: outb[m*N+n] = bf16( sigmoid(d) * aux[m*N+n] )
__global__ __launch_bounds__(256)
void wmma_gemm_kernel(const __bf16* __restrict__ A, const __bf16* __restrict__ W,
                      int M, int N, int K, int ldw, int mode,
                      float* __restrict__ outf, const float* __restrict__ bias,
                      const float* __restrict__ aux, __bf16* __restrict__ outb) {
    int gw = blockIdx.x * (blockDim.x >> 5) + (threadIdx.x >> 5);
    int ngrp = N >> 6;                 // groups of 4 N-tiles (N % 64 == 0)
    int groups = (M >> 4) * ngrp;
    if (gw >= groups) return;
    int mt = gw / ngrp, ng = gw - mt * ngrp;
    int m0 = mt << 4, n0 = ng << 6;

    v8f acc0 = {}, acc1 = {}, acc2 = {}, acc3 = {};
    for (int k0 = 0; k0 < K; k0 += 32) {
        v16bf a  = load_frag_A(A, K, m0, k0);
        v16bf b0 = load_frag_B(W, ldw, n0,      k0);
        v16bf b1 = load_frag_B(W, ldw, n0 + 16, k0);
        v16bf b2 = load_frag_B(W, ldw, n0 + 32, k0);
        v16bf b3 = load_frag_B(W, ldw, n0 + 48, k0);
        acc0 = wmma_bf16(a, b0, acc0);
        acc1 = wmma_bf16(a, b1, acc1);
        acc2 = wmma_bf16(a, b2, acc2);
        acc3 = wmma_bf16(a, b3, acc3);
    }

    int lane = threadIdx.x & 31;
    int n = lane & 15, half = (lane >> 4) & 1;
#pragma unroll
    for (int t = 0; t < 4; ++t) {
        v8f acc = (t == 0) ? acc0 : (t == 1) ? acc1 : (t == 2) ? acc2 : acc3;
        int nn = n0 + t * 16 + n;
#pragma unroll
        for (int r = 0; r < 8; ++r) {
            int m = m0 + r + half * 8;
            float d = acc[r];
            size_t idx = (size_t)m * N + nn;
            if (mode == 0) {
                outf[idx] = d;
            } else if (mode == 1) {
                outf[idx] = d + bias[nn];
            } else {
                float s = 1.0f / (1.0f + __expf(-d));
                outb[idx] = (__bf16)(s * aux[idx]);
            }
        }
    }
}

// ---------------- fused attention: scores -> softmax -> context ----------------
// block = one (p,b); 256 threads (8 waves). scores[q] = sum_h v[h]*tanh(Wq[q,b,h]+Wp[p,b,h])
__global__ __launch_bounds__(256)
void attn_kernel(const float* __restrict__ Wp,    // (P*B, H)
                 const float* __restrict__ Wq,    // (Q*B, H)
                 const float* __restrict__ vv,    // (H)
                 const float* __restrict__ question, // (Q*B, E) f32
                 const float* __restrict__ passage,  // (P*B, E) f32
                 __bf16* __restrict__ catb,       // (P*B, 2E) bf16
                 float* __restrict__ cf) {        // (P*B, E)  f32
    __shared__ float sc[QQ];
    __shared__ float aw[QQ];
    int p = blockIdx.x >> 4;
    int b = blockIdx.x & 15;
    int tid = threadIdx.x;
    int wave = tid >> 5, lane = tid & 31;

    const float* wp = Wp + (size_t)(p * BB + b) * HH;
    // each wave: 8 q values
    for (int q = wave * 8; q < wave * 8 + 8; ++q) {
        const float* wq = Wq + (size_t)(q * BB + b) * HH;
        float t = 0.f;
        for (int h = lane; h < HH; h += 32)
            t += vv[h] * tanhf(wq[h] + wp[h]);
#pragma unroll
        for (int off = 16; off > 0; off >>= 1) t += __shfl_xor(t, off, 32);
        if (lane == 0) sc[q] = t;
    }
    __syncthreads();
    if (tid == 0) {
        float mx = sc[0];
        for (int q = 1; q < QQ; ++q) mx = fmaxf(mx, sc[q]);
        float s = 0.f;
        for (int q = 0; q < QQ; ++q) { float e = __expf(sc[q] - mx); aw[q] = e; s += e; }
        float inv = 1.0f / s;
        for (int q = 0; q < QQ; ++q) aw[q] *= inv;
    }
    __syncthreads();
    // context: thread t owns e = t
    int e = tid;
    float c = 0.f;
    for (int q = 0; q < QQ; ++q)
        c += aw[q] * question[(size_t)(q * BB + b) * EE + e];
    size_t row = (size_t)p * BB + b;
    cf[row * EE + e] = c;
    catb[row * (2 * EE) + e]      = (__bf16)passage[row * EE + e];
    catb[row * (2 * EE) + EE + e] = (__bf16)c;
}

// ---------------- persistent sequential GRU ----------------
// one workgroup, 512 threads = 16 waves, single resident block (max VGPRs).
// Each wave OWNS its w_hh slice in registers: 3 N-tiles x 8 K-frags = 192 VGPRs.
// Per step: gh(16x768) = h(16x256) @ w_hh^T -> 24 WMMAs/wave, LDS-only A operands.
__global__ __launch_bounds__(512, 1)
void gru_kernel(const float* __restrict__ gi,    // (P*B, 3O), already + b_ih
                const __bf16* __restrict__ whh,  // (3O, O) bf16
                const float* __restrict__ bhh,   // (3O)
                float* __restrict__ out) {       // (P*B, O)
    extern __shared__ char smem[];
    float*  gh = (float*)smem;                           // 16*768*4 = 49152 B
    float*  hS = (float*)(smem + 49152);                 // 16*256*4 = 16384 B
    __bf16* hB = (__bf16*)(smem + 49152 + 16384);        // 16*256*2 =  8192 B

    int tid = threadIdx.x;
    int wave = tid >> 5;
    int t0 = wave, t1 = wave + 16, t2 = wave + 32;       // N-tiles (16 gate cols each)

    // preload this wave's w_hh fragments into registers (stays for all 512 steps)
    v16bf bw0[8], bw1[8], bw2[8];
#pragma unroll
    for (int kk = 0; kk < 8; ++kk) {
        bw0[kk] = load_frag_B(whh, OO, t0 * 16, kk * 32);
        bw1[kk] = load_frag_B(whh, OO, t1 * 16, kk * 32);
        bw2[kk] = load_frag_B(whh, OO, t2 * 16, kk * 32);
    }

    for (int i = tid; i < BB * OO; i += 512) { hS[i] = 0.f; hB[i] = (__bf16)0.0f; }
    __syncthreads();

    int lane = tid & 31;
    int n = lane & 15, half = (lane >> 4) & 1;

    for (int p = 0; p < PP; ++p) {
        // ---- gh = hB(16x256) @ whh^T : register-resident B, LDS A ----
        v8f a0 = {}, a1 = {}, a2 = {};
#pragma unroll
        for (int kk = 0; kk < 8; ++kk) {
            v16bf af = load_frag_A(hB, OO, 0, kk * 32);
            a0 = wmma_bf16(af, bw0[kk], a0);
            a1 = wmma_bf16(af, bw1[kk], a1);
            a2 = wmma_bf16(af, bw2[kk], a2);
        }
#pragma unroll
        for (int r = 0; r < 8; ++r) {
            int m = r + half * 8;
            gh[m * 768 + t0 * 16 + n] = a0[r];
            gh[m * 768 + t1 * 16 + n] = a1[r];
            gh[m * 768 + t2 * 16 + n] = a2[r];
        }
        __syncthreads();

        // ---- gate math + h update ----
        const float* gip = gi + (size_t)p * BB * 768;
        for (int idx = tid; idx < BB * OO; idx += 512) {
            int b = idx >> 8, o = idx & 255;
            float gr = gip[b * 768 + o]       + gh[b * 768 + o]       + bhh[o];
            float gz = gip[b * 768 + 256 + o] + gh[b * 768 + 256 + o] + bhh[256 + o];
            float gn = gip[b * 768 + 512 + o];
            float hn = gh[b * 768 + 512 + o] + bhh[512 + o];
            float r = 1.0f / (1.0f + __expf(-gr));
            float z = 1.0f / (1.0f + __expf(-gz));
            float nn = tanhf(gn + r * hn);
            float h = hS[idx];
            float hnew = (1.0f - z) * nn + z * h;
            hS[idx] = hnew;
            hB[idx] = (__bf16)hnew;
            out[((size_t)p * BB + b) * OO + o] = hnew;
        }
        __syncthreads();
    }
}

// ---------------- host launch ----------------
extern "C" void kernel_launch(void* const* d_in, const int* in_sizes, int n_in,
                              void* d_out, int out_size, void* d_ws, size_t ws_size,
                              hipStream_t stream) {
    const float* passage  = (const float*)d_in[0];   // (P,B,E)
    const float* question = (const float*)d_in[1];   // (Q,B,E)
    const float* Wuq      = (const float*)d_in[2];   // (H,E)
    const float* Wup      = (const float*)d_in[3];   // (H,E)
    const float* vvec     = (const float*)d_in[4];   // (1,H)
    const float* Wg       = (const float*)d_in[5];   // (2E,2E)
    const float* w_ih     = (const float*)d_in[6];   // (3O,E)
    const float* w_hh     = (const float*)d_in[7];   // (3O,O)
    const float* b_ih     = (const float*)d_in[8];   // (3O)
    const float* b_hh     = (const float*)d_in[9];   // (3O)
    float* out = (float*)d_out;                      // (P,B,O)

    // workspace carve-up
    char* ws = (char*)d_ws;
    size_t off = 0;
    auto alloc = [&](size_t bytes) { char* p = ws + off; off = (off + bytes + 255) & ~(size_t)255; return p; };

    float*  Wp_f    = (float*)alloc((size_t)PP * BB * HH * 4);       // 8 MB
    float*  Wq_f    = (float*)alloc((size_t)QQ * BB * HH * 4);       // 1 MB
    __bf16* pass_b  = (__bf16*)alloc((size_t)PP * BB * EE * 2);      // 4 MB
    __bf16* ques_b  = (__bf16*)alloc((size_t)QQ * BB * EE * 2);      // 0.5 MB
    __bf16* Wup_b   = (__bf16*)alloc((size_t)HH * EE * 2);
    __bf16* Wuq_b   = (__bf16*)alloc((size_t)HH * EE * 2);
    __bf16* Wg_b    = (__bf16*)alloc((size_t)2 * EE * 2 * EE * 2);   // 0.5 MB
    __bf16* wih_b   = (__bf16*)alloc((size_t)3 * OO * EE * 2);
    __bf16* whh_b   = (__bf16*)alloc((size_t)3 * OO * OO * 2);
    __bf16* cat_b   = (__bf16*)alloc((size_t)PP * BB * 2 * EE * 2);  // 8 MB
    float*  c_f     = (float*)alloc((size_t)PP * BB * EE * 4);       // 8 MB
    __bf16* x_b     = (__bf16*)alloc((size_t)PP * BB * EE * 2);      // 4 MB
    float*  gi_f    = (float*)alloc((size_t)PP * BB * 3 * OO * 4);   // 24 MB
    (void)ws_size;

    auto conv = [&](const float* src, __bf16* dst, int n) {
        f32_to_bf16_kernel<<<(n + 255) / 256, 256, 0, stream>>>(src, dst, n);
    };
    conv(passage, pass_b, PP * BB * EE);
    conv(question, ques_b, QQ * BB * EE);
    conv(Wup, Wup_b, HH * EE);
    conv(Wuq, Wuq_b, HH * EE);
    conv(Wg, Wg_b, 2 * EE * 2 * EE);
    conv(w_ih, wih_b, 3 * OO * EE);
    conv(w_hh, whh_b, 3 * OO * OO);

    auto gemm = [&](const __bf16* A, const __bf16* W, int M, int N, int K, int ldw,
                    int mode, float* outf, const float* bias, const float* aux, __bf16* outb) {
        int groups = (M / 16) * (N / 64);   // one wave per 16x64 strip
        int blocks = (groups + 7) / 8;      // 8 waves per 256-thread block
        wmma_gemm_kernel<<<blocks, 256, 0, stream>>>(A, W, M, N, K, ldw, mode,
                                                     outf, bias, aux, outb);
    };

    // Wp = passage @ Wup^T ; Wq = question @ Wuq^T
    gemm(pass_b, Wup_b, PP * BB, HH, EE, EE, 0, Wp_f, nullptr, nullptr, nullptr);
    gemm(ques_b, Wuq_b, QQ * BB, HH, EE, EE, 0, Wq_f, nullptr, nullptr, nullptr);

    // fused attention -> cat (bf16) and c (f32)
    attn_kernel<<<PP * BB, 256, 0, stream>>>(Wp_f, Wq_f, vvec, question, passage,
                                             cat_b, c_f);

    // gated input: x = sigmoid(cat @ Wg[E:,:]^T) * c   (only the used half of Wg)
    gemm(cat_b, Wg_b + (size_t)EE * (2 * EE), PP * BB, EE, 2 * EE, 2 * EE,
         2, nullptr, nullptr, c_f, x_b);

    // gi = x @ w_ih^T + b_ih
    gemm(x_b, wih_b, PP * BB, 3 * OO, EE, EE, 1, gi_f, b_ih, nullptr, nullptr);

    // sequential GRU: single persistent workgroup, 16 waves, 72KB dynamic LDS
    gru_kernel<<<1, 512, 49152 + 16384 + 8192, stream>>>(gi_f, whh_b, b_hh, out);

    (void)in_sizes; (void)n_in; (void)out_size;
}
